// SelfAttentionMulti_23467701305905
// MI455X (gfx1250) — compile-verified, matrix operand order checked by reading, not run on hardware
//
#include <hip/hip_runtime.h>

// ---------------------------------------------------------------------------
// Multi-head self-attention for MI455X (gfx1250), bf16 WMMA path.
//   B=8, N=1024, D=768, H=12, Hd=64.
// Pipeline:
//   0) elementwise fp32->bf16 convert of x and all weights (once, HBM-bound)
//   1) fused QKV projection (pure bf16 WMMA GEMM, 32x32 wave tile)
//   2) flash attention: 64-key chunks, block-shared K/V staging in LDS via
//      async load-to-LDS, online softmax, LDS C->A transpose of P
//   3) output projection (bf16 WMMA GEMM, 32x64 wave tile, fp32 out)
// Workspace (u16 elements):
//   qbuf [B,H,N,Hd] | kbuf [B,H,N,Hd] | vbuf [B,H,Hd,N] (transposed)
//   abuf [B*N, D]   | xbf [B*N, D]    | wqb/wkb/wvb/wob [D, D]
// Total ~67.7 MB.
// ---------------------------------------------------------------------------

#define BATCH 8
#define SEQ   1024
#define DIM   768
#define NHEAD 12
#define HDIM  64

typedef unsigned short u16;
typedef __attribute__((ext_vector_type(8)))  u16    u16x8;
typedef __attribute__((ext_vector_type(16))) u16    u16x16;
typedef __attribute__((ext_vector_type(4)))  int    i32x4;
typedef __attribute__((ext_vector_type(2)))  __bf16 bf16x2;
typedef __attribute__((ext_vector_type(16))) __bf16 bf16x16;
typedef __attribute__((ext_vector_type(8)))  float  f32x8;

// ---- fp32 -> bf16 conversion helpers --------------------------------------
static __device__ __forceinline__ u16 cvt_bf16(float f) {
  unsigned int u = __float_as_uint(f);
  unsigned int r = u + 0x7fffu + ((u >> 16) & 1u);
  return (u16)(r >> 16);
}

#if __has_builtin(__builtin_amdgcn_cvt_pk_bf16_f32)
static __device__ __forceinline__ unsigned int cvt2_bf16(float a, float b) {
  bf16x2 p = __builtin_amdgcn_cvt_pk_bf16_f32(a, b);
  return __builtin_bit_cast(unsigned int, p);
}
#else
static __device__ __forceinline__ unsigned int cvt2_bf16(float a, float b) {
  return (unsigned int)cvt_bf16(a) | ((unsigned int)cvt_bf16(b) << 16);
}
#endif

static __device__ __forceinline__ u16x8 cvt8_bf16(f32x8 x) {
  uint4 w;
  w.x = cvt2_bf16(x[0], x[1]);
  w.y = cvt2_bf16(x[2], x[3]);
  w.z = cvt2_bf16(x[4], x[5]);
  w.w = cvt2_bf16(x[6], x[7]);
  return __builtin_bit_cast(u16x8, w);
}

// ---- async global -> LDS staging (CDNA5 ASYNCcnt path), with fallback -----
#if __has_builtin(__builtin_amdgcn_global_load_async_to_lds_b128)
#define HAVE_ASYNC_LDS 1
static __device__ __forceinline__ void stage16(const u16* g, u16* l) {
  __builtin_amdgcn_global_load_async_to_lds_b128((i32x4*)g, (i32x4*)l, 0, 0);
}
static __device__ __forceinline__ void stage_wait() {
#if __has_builtin(__builtin_amdgcn_s_wait_asynccnt)
  __builtin_amdgcn_s_wait_asynccnt(0);
#else
  asm volatile("s_wait_asynccnt 0" ::: "memory");
#endif
}
#else
static __device__ __forceinline__ void stage16(const u16* g, u16* l) {
  *(u16x8*)l = *(const u16x8*)g;
}
static __device__ __forceinline__ void stage_wait() {}
#endif

static __device__ __forceinline__ f32x8 wmma_bf16(u16x16 a, u16x16 b, f32x8 c) {
  return __builtin_amdgcn_wmma_f32_16x16x32_bf16(
      /*neg_a=*/false, __builtin_bit_cast(bf16x16, a),
      /*neg_b=*/false, __builtin_bit_cast(bf16x16, b),
      /*c_mod=*/(short)0, c, /*reuse_a=*/false, /*reuse_b=*/false);
}

// A-operand (16x32 bf16) from a bf16(u16) row-major matrix (global or LDS).
// lane&15 = row, lane>>4 selects K-chunks {8h..8h+7} and {16+8h..16+8h+7}.
// (B-operand of D=A*B is the A-layout of B^T, i.e. rows of W for y = x@W^T.)
static __device__ __forceinline__ u16x16
load_a_bf16(const u16* base, int ld, int row, int k0, int lane) {
  const int m = lane & 15, hf = lane >> 4;
  const u16* p = base + (size_t)(row + m) * ld + k0 + 8 * hf;
  u16x8 a = *(const u16x8*)(p);
  u16x8 b = *(const u16x8*)(p + 16);
  u16x16 r;
#pragma unroll
  for (int i = 0; i < 8; ++i) { r[i] = a[i]; r[8 + i] = b[i]; }
  return r;
}

// ---------------------------------------------------------------------------
// Kernel 0: elementwise fp32 -> bf16 (8 elements / thread, b128 in, b128 out)
// ---------------------------------------------------------------------------
__global__ void __launch_bounds__(256)
cvt_f32_bf16_kernel(const float* __restrict__ src, u16* __restrict__ dst,
                    int n8) {
  const int i = blockIdx.x * 256 + threadIdx.x;
  if (i >= n8) return;
  const float4 a = *(const float4*)(src + (size_t)i * 8);
  const float4 b = *(const float4*)(src + (size_t)i * 8 + 4);
  uint4 w;
  w.x = cvt2_bf16(a.x, a.y);
  w.y = cvt2_bf16(a.z, a.w);
  w.z = cvt2_bf16(b.x, b.y);
  w.w = cvt2_bf16(b.z, b.w);
  *(uint4*)(dst + (size_t)i * 8) = w;
}

// ---------------------------------------------------------------------------
// Kernel 1: fused QKV projection. y = x @ W^T + b  (B-operand == rows of W).
// Wave tile: 32 rows x 32 features for each of Q, K, V (12 accum tiles).
// grid = (8192/32/8, 768/32), block = 256 (8 waves).
// ---------------------------------------------------------------------------
__global__ void __launch_bounds__(256)
qkv_proj_kernel(const u16* __restrict__ xbf,
                const u16* __restrict__ wqb, const float* __restrict__ bq,
                const u16* __restrict__ wkb, const float* __restrict__ bk,
                const u16* __restrict__ wvb, const float* __restrict__ bv,
                u16* __restrict__ qbuf, u16* __restrict__ kbuf,
                u16* __restrict__ vbuf) {
  const int lane = threadIdx.x & 31;
  const int wave = threadIdx.x >> 5;
  const int m0 = (blockIdx.x * 8 + wave) * 32;   // row block in [0, 8192)
  const int n0 = blockIdx.y * 32;                // feature block in [0, 768)
  const int hf = lane >> 4;
  const int nc = lane & 15;

  f32x8 aq[2][2], ak[2][2], av[2][2];            // [m-subtile][n-subtile]
#pragma unroll
  for (int s = 0; s < 2; ++s) {
    const int f = n0 + 16 * s + nc;
    const float vq = bq[f], vk = bk[f], vv = bv[f];
#pragma unroll
    for (int u = 0; u < 2; ++u)
#pragma unroll
      for (int i = 0; i < 8; ++i) {
        aq[u][s][i] = vq; ak[u][s][i] = vk; av[u][s][i] = vv;
      }
  }

  for (int k0 = 0; k0 < DIM; k0 += 32) {
    u16x16 A0 = load_a_bf16(xbf, DIM, m0,      k0, lane);
    u16x16 A1 = load_a_bf16(xbf, DIM, m0 + 16, k0, lane);
#pragma unroll
    for (int s = 0; s < 2; ++s) {
      const int nr = n0 + 16 * s;
      u16x16 Bq = load_a_bf16(wqb, DIM, nr, k0, lane);
      aq[0][s] = wmma_bf16(A0, Bq, aq[0][s]);
      aq[1][s] = wmma_bf16(A1, Bq, aq[1][s]);
      u16x16 Bk = load_a_bf16(wkb, DIM, nr, k0, lane);
      ak[0][s] = wmma_bf16(A0, Bk, ak[0][s]);
      ak[1][s] = wmma_bf16(A1, Bk, ak[1][s]);
      u16x16 Bv = load_a_bf16(wvb, DIM, nr, k0, lane);
      av[0][s] = wmma_bf16(A0, Bv, av[0][s]);
      av[1][s] = wmma_bf16(A1, Bv, av[1][s]);
    }
  }

  const float scale = rsqrtf((float)DIM);  // reference scales by embed_dim^-0.5
#pragma unroll
  for (int u = 0; u < 2; ++u) {
    const int mt = m0 + 16 * u;
#pragma unroll
    for (int s = 0; s < 2; ++s) {
      const int f  = n0 + 16 * s + nc;
      const int hh = f / HDIM;
      const int hd = f % HDIM;
      const u16x8 pq = cvt8_bf16(aq[u][s] * scale);
      const u16x8 pk = cvt8_bf16(ak[u][s]);
      // Q (scaled) and K: [b, h, n, hd] bf16, scattered u16 stores.
#pragma unroll
      for (int v = 0; v < 8; ++v) {
        const int m   = mt + v + 8 * hf;
        const int bi  = m >> 10;
        const int tok = m & (SEQ - 1);
        const size_t qi = (((size_t)bi * NHEAD + hh) * SEQ + tok) * HDIM + hd;
        qbuf[qi] = pq[v];
        kbuf[qi] = pk[v];
      }
      // V transposed: [b, h, hd, n] bf16 -- 8 consecutive tokens -> one b128.
      const u16x8 pv = cvt8_bf16(av[u][s]);
      const int mb  = mt + 8 * hf;
      const int bi  = mb >> 10;
      const int tok = mb & (SEQ - 1);
      const size_t vi = (((size_t)bi * NHEAD + hh) * HDIM + hd) * SEQ + tok;
      *(u16x8*)(vbuf + vi) = pv;
    }
  }
}

// ---------------------------------------------------------------------------
// Kernel 2: flash attention. 8 waves/block share one (b,h); each wave owns a
// 16-query tile. Per 64-key chunk the block cooperatively stages K (64x64)
// and V^T (64x64) into LDS with async load-to-LDS, then each wave runs
// scores -> online softmax -> P@V with WMMA operands read from LDS.
// grid = (B*H, SEQ/16/8), block = 256 (8 waves).
// ---------------------------------------------------------------------------
__global__ void __launch_bounds__(256)
flash_attn_kernel(const u16* __restrict__ qbuf, const u16* __restrict__ kbuf,
                  const u16* __restrict__ vbuf, u16* __restrict__ abuf) {
  __shared__ u16 kTile[64 * HDIM];   // 8 KB: [key][hd]
  __shared__ u16 vTile[HDIM * 64];   // 8 KB: [hd][key]
  __shared__ u16 ldsP[8][16 * 64];   // 16 KB: per-wave P transpose scratch
  const int tid  = threadIdx.x;
  const int lane = tid & 31;
  const int wave = tid >> 5;
  const int bh = blockIdx.x;                    // 0..95
  const int b  = bh / NHEAD;
  const int h  = bh % NHEAD;
  const int q0 = (blockIdx.y * 8 + wave) * 16;  // query tile base
  const int hf = lane >> 4;
  const int nc = lane & 15;

  const u16* qp = qbuf + (size_t)bh * SEQ * HDIM;
  const u16* kp = kbuf + (size_t)bh * SEQ * HDIM;
  const u16* vp = vbuf + (size_t)bh * HDIM * SEQ;
  u16* lp = &ldsP[wave][0];

  // cooperative staging: 256 threads x 16 u16 (2 x b128) per 64x64 tile
  const int sRow = tid >> 2;             // 0..63
  const int sCol = (tid & 3) * 16;       // 0,16,32,48
  const u16* kg = kp + (size_t)sRow * HDIM + sCol;
  const u16* vg = vp + (size_t)sRow * SEQ + sCol;
  u16* kl = &kTile[sRow * HDIM + sCol];
  u16* vl = &vTile[sRow * 64 + sCol];

  // Q A-operands are loop-invariant (K = hd, two 32-chunks).
  const u16x16 qa0 = load_a_bf16(qp, HDIM, q0, 0, lane);
  const u16x16 qa1 = load_a_bf16(qp, HDIM, q0, 32, lane);

  f32x8 acc[4];
  float mrow[8], lrow[8];
#pragma unroll
  for (int t = 0; t < 4; ++t)
#pragma unroll
    for (int i = 0; i < 8; ++i) acc[t][i] = 0.0f;
#pragma unroll
  for (int i = 0; i < 8; ++i) { mrow[i] = -__builtin_inff(); lrow[i] = 0.0f; }

  for (int c = 0; c < SEQ / 64; ++c) {
    const int kk = c * 64;
    // ---- stage K/V chunk into LDS (ASYNCcnt path on CDNA5) ----
    stage16(kg + (size_t)kk * HDIM, kl);
    stage16(kg + (size_t)kk * HDIM + 8, kl + 8);
    stage16(vg + kk, vl);
    stage16(vg + kk + 8, vl + 8);
    stage_wait();
    __syncthreads();

    // ---- scores: four 16x16 tiles over hd (K = 64 -> 2 WMMAs each) ----
    f32x8 s[4];
#pragma unroll
    for (int j = 0; j < 4; ++j) {
#pragma unroll
      for (int i = 0; i < 8; ++i) s[j][i] = 0.0f;
      s[j] = wmma_bf16(qa0, load_a_bf16(kTile, HDIM, 16 * j, 0,  lane), s[j]);
      s[j] = wmma_bf16(qa1, load_a_bf16(kTile, HDIM, 16 * j, 32, lane), s[j]);
    }

    // ---- online softmax (row = VGPR idx + 8*half; reduce across 16 lanes) --
#pragma unroll
    for (int v = 0; v < 8; ++v) {
      float mx = fmaxf(fmaxf(s[0][v], s[1][v]), fmaxf(s[2][v], s[3][v]));
      mx = fmaxf(mx, __shfl_xor(mx, 1, 32));
      mx = fmaxf(mx, __shfl_xor(mx, 2, 32));
      mx = fmaxf(mx, __shfl_xor(mx, 4, 32));
      mx = fmaxf(mx, __shfl_xor(mx, 8, 32));
      const float nm   = fmaxf(mrow[v], mx);
      const float corr = __expf(mrow[v] - nm);
      mrow[v] = nm;
      const float p0 = __expf(s[0][v] - nm);
      const float p1 = __expf(s[1][v] - nm);
      const float p2 = __expf(s[2][v] - nm);
      const float p3 = __expf(s[3][v] - nm);
      s[0][v] = p0; s[1][v] = p1; s[2][v] = p2; s[3][v] = p3;
      float rs = (p0 + p1) + (p2 + p3);
      rs += __shfl_xor(rs, 1, 32);
      rs += __shfl_xor(rs, 2, 32);
      rs += __shfl_xor(rs, 4, 32);
      rs += __shfl_xor(rs, 8, 32);
      lrow[v] = lrow[v] * corr + rs;
      acc[0][v] *= corr; acc[1][v] *= corr;
      acc[2][v] *= corr; acc[3][v] *= corr;
    }

    // ---- C-layout -> A-layout transpose of P through LDS (16x64 bf16) ----
#pragma unroll
    for (int v = 0; v < 8; ++v) {
      const int row = v + 8 * hf;
      const unsigned int w0 = cvt2_bf16(s[0][v], s[1][v]);  // v_cvt_pk
      const unsigned int w1 = cvt2_bf16(s[2][v], s[3][v]);
      lp[row * 64 + nc]      = (u16)(w0 & 0xffffu);
      lp[row * 64 + 16 + nc] = (u16)(w0 >> 16);
      lp[row * 64 + 32 + nc] = (u16)(w1 & 0xffffu);
      lp[row * 64 + 48 + nc] = (u16)(w1 >> 16);
    }
    asm volatile("s_wait_dscnt 0" ::: "memory");
    const u16x16 pa0 = load_a_bf16(lp, 64, 0, 0,  lane);
    const u16x16 pa1 = load_a_bf16(lp, 64, 0, 32, lane);

    // ---- acc += P (16x64) @ V (64x64); B-op = rows of V^T in LDS ----
#pragma unroll
    for (int t = 0; t < 4; ++t) {
      acc[t] = wmma_bf16(pa0, load_a_bf16(vTile, 64, 16 * t, 0,  lane), acc[t]);
      acc[t] = wmma_bf16(pa1, load_a_bf16(vTile, 64, 16 * t, 32, lane), acc[t]);
    }

    __syncthreads();   // protect kTile/vTile before next chunk's staging
  }

  // ---- normalize and store attention output [B*N, D] as bf16 ----
  float inv[8];
#pragma unroll
  for (int v = 0; v < 8; ++v) inv[v] = 1.0f / lrow[v];
#pragma unroll
  for (int t = 0; t < 4; ++t) {
    f32x8 o;
#pragma unroll
    for (int v = 0; v < 8; ++v) o[v] = acc[t][v] * inv[v];
    const u16x8 pk = cvt8_bf16(o);
#pragma unroll
    for (int v = 0; v < 8; ++v) {
      const int tok = q0 + v + 8 * hf;
      abuf[((size_t)b * SEQ + tok) * DIM + h * HDIM + t * 16 + nc] = pk[v];
    }
  }
}

// ---------------------------------------------------------------------------
// Kernel 3: output projection. out = attn @ Wo^T + bo  (fp32 out).
// Wave tile: 32 rows x 64 features. grid = (8192/32/8, 768/64), block = 256.
// ---------------------------------------------------------------------------
__global__ void __launch_bounds__(256)
out_proj_kernel(const u16* __restrict__ abuf, const u16* __restrict__ wob,
                const float* __restrict__ bo, float* __restrict__ out) {
  const int lane = threadIdx.x & 31;
  const int wave = threadIdx.x >> 5;
  const int m0 = (blockIdx.x * 8 + wave) * 32;
  const int n0 = blockIdx.y * 64;
  const int hf = lane >> 4;
  const int nc = lane & 15;

  f32x8 acc[2][4];
#pragma unroll
  for (int t = 0; t < 4; ++t) {
    const float bv = bo[n0 + 16 * t + nc];
#pragma unroll
    for (int u = 0; u < 2; ++u)
#pragma unroll
      for (int i = 0; i < 8; ++i) acc[u][t][i] = bv;
  }

  for (int k0 = 0; k0 < DIM; k0 += 32) {
    u16x16 A0 = load_a_bf16(abuf, DIM, m0,      k0, lane);
    u16x16 A1 = load_a_bf16(abuf, DIM, m0 + 16, k0, lane);
#pragma unroll
    for (int t = 0; t < 4; ++t) {
      u16x16 Bt = load_a_bf16(wob, DIM, n0 + 16 * t, k0, lane);
      acc[0][t] = wmma_bf16(A0, Bt, acc[0][t]);
      acc[1][t] = wmma_bf16(A1, Bt, acc[1][t]);
    }
  }

#pragma unroll
  for (int u = 0; u < 2; ++u)
#pragma unroll
    for (int t = 0; t < 4; ++t) {
      const int col = n0 + 16 * t + nc;
#pragma unroll
      for (int v = 0; v < 8; ++v)
        out[(size_t)(m0 + 16 * u + v + 8 * hf) * DIM + col] = acc[u][t][v];
    }
}

// ---------------------------------------------------------------------------
extern "C" void kernel_launch(void* const* d_in, const int* in_sizes, int n_in,
                              void* d_out, int out_size, void* d_ws,
                              size_t ws_size, hipStream_t stream) {
  (void)in_sizes; (void)n_in; (void)out_size; (void)ws_size;
  const float* x  = (const float*)d_in[0];
  const float* Wq = (const float*)d_in[1];
  const float* bq = (const float*)d_in[2];
  const float* Wk = (const float*)d_in[3];
  const float* bk = (const float*)d_in[4];
  const float* Wv = (const float*)d_in[5];
  const float* bv = (const float*)d_in[6];
  const float* Wo = (const float*)d_in[7];
  const float* bo = (const float*)d_in[8];
  float* out = (float*)d_out;

  const size_t elems_x = (size_t)BATCH * SEQ * DIM;  // 6,291,456
  const size_t elems_w = (size_t)DIM * DIM;          //   589,824
  u16* qbuf = (u16*)d_ws;
  u16* kbuf = qbuf + elems_x;
  u16* vbuf = kbuf + elems_x;
  u16* abuf = vbuf + elems_x;
  u16* xbf  = abuf + elems_x;
  u16* wqb  = xbf + elems_x;
  u16* wkb  = wqb + elems_w;
  u16* wvb  = wkb + elems_w;
  u16* wob  = wvb + elems_w;

  dim3 blk(256, 1, 1);
  // 0) convert inputs to bf16 once
  const int nx8 = (int)(elems_x / 8);  // 786,432
  const int nw8 = (int)(elems_w / 8);  //  73,728
  cvt_f32_bf16_kernel<<<dim3((nx8 + 255) / 256), blk, 0, stream>>>(x, xbf, nx8);
  cvt_f32_bf16_kernel<<<dim3((nw8 + 255) / 256), blk, 0, stream>>>(Wq, wqb, nw8);
  cvt_f32_bf16_kernel<<<dim3((nw8 + 255) / 256), blk, 0, stream>>>(Wk, wkb, nw8);
  cvt_f32_bf16_kernel<<<dim3((nw8 + 255) / 256), blk, 0, stream>>>(Wv, wvb, nw8);
  cvt_f32_bf16_kernel<<<dim3((nw8 + 255) / 256), blk, 0, stream>>>(Wo, wob, nw8);

  // 1) QKV projection
  dim3 g1(32, 24, 1);   // 256 m-tiles(32) / 8 waves, 768/32 feature blocks
  qkv_proj_kernel<<<g1, blk, 0, stream>>>(xbf, wqb, bq, wkb, bk, wvb, bv,
                                          qbuf, kbuf, vbuf);
  // 2) attention
  dim3 g2(BATCH * NHEAD, SEQ / (16 * 8), 1);  // (96, 8)
  flash_attn_kernel<<<g2, blk, 0, stream>>>(qbuf, kbuf, vbuf, abuf);
  // 3) output projection
  dim3 g3(32, 12, 1);   // 256 m-tiles(32) / 8 waves, 768/64 feature blocks
  out_proj_kernel<<<g3, blk, 0, stream>>>(abuf, wob, bo, out);
}
